// PB_RPA_77635828842945
// MI455X (gfx1250) — matrix operand order repaired and checked by reading, not compiled, not run on hardware
//
#include <hip/hip_runtime.h>
#include <hip/hip_bf16.h>

// ---------------- problem constants ----------------
constexpr int NHh   = 16;
constexpr int Dd    = 1024;
constexpr int Bb    = 4;
constexpr int Tt    = 4096;
constexpr int HD    = 64;          // Dd / NHh
constexpr int MM    = Bb * Tt;     // 16384 GEMM rows
constexpr int CHUNK = 256;         // scan chunk along T
constexpr int NC    = Tt / CHUNK;  // 16 chunks

// ---------------- vector types for WMMA ----------------
typedef __attribute__((ext_vector_type(16))) __bf16 v16bf;
typedef __attribute__((ext_vector_type(8)))  __bf16 v8bf;
typedef __attribute__((ext_vector_type(8)))  float  v8f;

// ---------------- helpers ----------------
__device__ __forceinline__ unsigned short f2bf(float f) {
    union { float f; unsigned int u; } v; v.f = f;
    unsigned int u = v.u;
    unsigned int r = u + 0x7fffu + ((u >> 16) & 1u);   // round-to-nearest-even
    return (unsigned short)(r >> 16);
}

// Async global->LDS copy of 32 bytes per lane (two b128 beats).
// INST_OFFSET is added to BOTH the LDS destination and the global address
// (CDNA5 ISA), so one address pair covers both beats.
__device__ __forceinline__ void async_ld32B(unsigned lds_off, const void* gaddr) {
    asm volatile("global_load_async_to_lds_b128 %0, %1, off\n\t"
                 "global_load_async_to_lds_b128 %0, %1, off offset:16"
                 :: "v"(lds_off), "v"(gaddr) : "memory");
}
__device__ __forceinline__ void wait_async0() {
    asm volatile("s_wait_asynccnt 0x0" ::: "memory");
}

// ---------------- f32 -> bf16 conversion (4 elems/thread) ----------------
__global__ __launch_bounds__(256) void k_f32_to_bf16_x4(const float* __restrict__ in,
                                                        unsigned short* __restrict__ out,
                                                        int n4) {
    int i = blockIdx.x * 256 + threadIdx.x;
    if (i >= n4) return;
    const float4 f = ((const float4*)in)[i];
    uint2 p;
    p.x = (unsigned)f2bf(f.x) | ((unsigned)f2bf(f.y) << 16);
    p.y = (unsigned)f2bf(f.z) | ((unsigned)f2bf(f.w) << 16);
    ((uint2*)out)[i] = p;
}

// ---------------- fused convert + transpose: W(NxK,f32) -> Wt(KxN,bf16) ----------------
__global__ __launch_bounds__(256) void k_transpose_to_bf16(const float* __restrict__ Win,
                                                           unsigned short* __restrict__ Wt) {
    __shared__ unsigned short tile[32][33];   // +1 pad vs bank conflicts
    const int tx = threadIdx.x & 31;
    const int ty = threadIdx.x >> 5;          // 0..7
    const int n0 = blockIdx.x * 32;
    const int k0 = blockIdx.y * 32;
#pragma unroll
    for (int r = 0; r < 32; r += 8) {
        float v = Win[(size_t)(n0 + ty + r) * Dd + k0 + tx];
        tile[ty + r][tx] = f2bf(v);
    }
    __syncthreads();
#pragma unroll
    for (int r = 0; r < 32; r += 8) {
        Wt[(size_t)(k0 + ty + r) * Dd + n0 + tx] = tile[tx][ty + r];
    }
}

// ---------------- bf16 WMMA GEMM: C(MxN,f32) = A(MxK) * Bt(KxN) ----------------
// Bt is the pre-transposed weight (so C = A @ W^T). K = N = 1024 fixed.
// Block tile 128x128, 8 waves, each wave 32(M)x64(N).
// LDS double-buffered; tiles staged with GLOBAL_LOAD_ASYNC_TO_LDS_B128 (ASYNCcnt).
__global__ __launch_bounds__(256) void gemm_bf16_nt(const unsigned short* __restrict__ A,
                                                    const unsigned short* __restrict__ Bt,
                                                    float* __restrict__ C, int M) {
    constexpr int K = 1024, N = 1024;
    __shared__ alignas(16) unsigned short ldsA[2][128][32];   // [buf][m][k]  8KiB/buf
    __shared__ alignas(32) unsigned short ldsB[2][32][128];   // [buf][k][n]  8KiB/buf

    const int tid  = threadIdx.x;
    const int lane = tid & 31;
    const int wave = tid >> 5;
    const int wm   = wave & 3;      // 0..3 -> M offset wm*32
    const int wn   = wave >> 2;     // 0..1 -> N offset wn*64
    const int m0   = blockIdx.y * 128;
    const int n0   = blockIdx.x * 128;
    const int g    = lane >> 4;     // half-wave select
    const int l16  = lane & 15;

    v8f acc[2][4] = {};

    const int arow = tid >> 1;           // 0..127 A staging row (m)
    const int akk  = (tid & 1) * 16;     // 0 / 16
    const int brow = tid >> 3;           // 0..31  B staging row (k)
    const int bcol = (tid & 7) * 16;     // 0..112 step 16

    const unsigned ldsA_off = (unsigned)(size_t)&ldsA[0][arow][akk];
    const unsigned ldsB_off = (unsigned)(size_t)&ldsB[0][brow][bcol];
    constexpr unsigned BUFSTRIDE = 128 * 32 * 2;   // 8 KiB per buffer (same for A and B)

    auto stage = [&](int buf, int k0) {
        async_ld32B(ldsA_off + buf * BUFSTRIDE,
                    A + (size_t)(m0 + arow) * K + k0 + akk);
        async_ld32B(ldsB_off + buf * BUFSTRIDE,
                    Bt + (size_t)(k0 + brow) * N + n0 + bcol);
    };

    stage(0, 0);

    for (int k0 = 0; k0 < K; k0 += 32) {
        const int cur = (k0 >> 5) & 1;
        wait_async0();          // own async writes into `cur` are in LDS
        __syncthreads();        // everyone's writes visible; buffer reuse safe
        if (k0 + 32 < K) {
            stage(cur ^ 1, k0 + 32);
            if (k0 + 64 < K) {  // pull tile after next toward L2
                __builtin_prefetch(A  + (size_t)(m0 + arow) * K + k0 + 64 + akk, 0, 0);
                __builtin_prefetch(Bt + (size_t)(k0 + 64 + brow) * N + n0 + bcol, 0, 0);
            }
        }

        // ---- A fragments (16-bit A layout: per-lane chunks at k = 8g and 16+8g) ----
        v16bf afrag[2];
#pragma unroll
        for (int mi = 0; mi < 2; ++mi) {
            int r = wm * 32 + mi * 16 + l16;
            v8bf c0 = *(const v8bf*)&ldsA[cur][r][8 * g];
            v8bf c1 = *(const v8bf*)&ldsA[cur][r][16 + 8 * g];
            afrag[mi] = __builtin_shufflevector(c0, c1,
                0,1,2,3,4,5,6,7,8,9,10,11,12,13,14,15);
        }
        // ---- B fragments: lane holds column n = l16 of B(KxN), k at 8g+s / 16+8g+(s-8) ----
#pragma unroll
        for (int ni = 0; ni < 4; ++ni) {
            v16bf bfrag;
            const __bf16* bp = (const __bf16*)&ldsB[cur][0][wn * 64 + ni * 16 + l16];
#pragma unroll
            for (int s = 0; s < 16; ++s)
                bfrag[s] = bp[(s < 8 ? 8 * g + s : 16 + 8 * g + (s - 8)) * 128];
#pragma unroll
            for (int mi = 0; mi < 2; ++mi) {
                acc[mi][ni] = __builtin_amdgcn_wmma_f32_16x16x32_bf16(
                    false, afrag[mi], false, bfrag, (short)0, acc[mi][ni], false, false);
            }
        }
    }

    // ---- store C (f32 C/D layout: VGPR r -> M = r + 8g, lane.l16 -> N) ----
#pragma unroll
    for (int mi = 0; mi < 2; ++mi) {
#pragma unroll
        for (int ni = 0; ni < 4; ++ni) {
            const int col  = n0 + wn * 64 + ni * 16 + l16;
            const int rowb = m0 + wm * 32 + mi * 16 + 8 * g;
#pragma unroll
            for (int r = 0; r < 8; ++r)
                C[(size_t)(rowb + r) * N + col] = acc[mi][ni][r];
        }
    }
}

// ---------------- phi: penalty[b][t] = (phi - cumsum(phi)/(t+1))^2 ----------------
__global__ __launch_bounds__(256) void k_phi(const float* __restrict__ phi,
                                             float* __restrict__ penalty) {
    const int b   = blockIdx.x;
    const int tid = threadIdx.x;
    const size_t base = (size_t)b * Tt + tid * 16;
    float o[16], v[16];
    float run = 0.f;
#pragma unroll
    for (int i = 0; i < 16; ++i) { o[i] = phi[base + i]; run += o[i]; v[i] = run; }
    __shared__ float sums[256];
    sums[tid] = run;
    __syncthreads();
    for (int off = 1; off < 256; off <<= 1) {
        float x = (tid >= off) ? sums[tid - off] : 0.f;
        __syncthreads();
        sums[tid] += x;
        __syncthreads();
    }
    const float exc = sums[tid] - run;
#pragma unroll
    for (int i = 0; i < 16; ++i) {
        int   t    = tid * 16 + i;
        float cum  = v[i] + exc;
        float mean = cum / (float)(t + 1);
        float dp   = o[i] - mean;
        penalty[base + i] = dp * dp;
    }
}

// ---------------- chunk partial sums of w^2 over T ----------------
__global__ __launch_bounds__(256) void k_partial_wsq(const float* __restrict__ w,
                                                     float* __restrict__ part) {
    const int d = blockIdx.x * 256 + threadIdx.x;
    const int c = blockIdx.y, b = blockIdx.z;
    const float* p = w + ((size_t)(b * Tt + c * CHUNK)) * Dd + d;
    float s = 0.f;
    for (int i = 0; i < CHUNK; ++i) { float v = p[(size_t)i * Dd]; s += v * v; }
    part[((size_t)b * NC + c) * Dd + d] = s;
}

// ---------------- exclusive prefix over chunks, per (b,d) ----------------
__global__ __launch_bounds__(256) void k_prefix(const float* __restrict__ part,
                                                float* __restrict__ off) {
    const int s = blockIdx.x * 256 + threadIdx.x;   // 0..B*D-1
    const int b = s / Dd, d = s % Dd;
    float run = 0.f;
    for (int c = 0; c < NC; ++c) {
        size_t ix = ((size_t)b * NC + c) * Dd + d;
        off[ix] = run;
        run += part[ix];
    }
}

// ---------------- score: running denom + tssa - gamma*penalty ----------------
// One wave32 per (chunk, head, batch): lane owns hd lanes dl and dl+32,
// 64-wide reduction = 1 add + 5 in-wave shuffles. No barriers.
__global__ __launch_bounds__(32) void k_score(const float* __restrict__ w,
                                              const float* __restrict__ off1,
                                              const float* __restrict__ penalty,
                                              const float* __restrict__ gamma,
                                              const float* __restrict__ temp,
                                              float* __restrict__ score) {
    const int c = blockIdx.x, h = blockIdx.y, b = blockIdx.z;
    const int dl = threadIdx.x;                 // 0..31
    const int d0 = h * HD + dl;
    const int d1 = d0 + 32;
    float run0 = off1[((size_t)b * NC + c) * Dd + d0];
    float run1 = off1[((size_t)b * NC + c) * Dd + d1];
    const float tmph = temp[h], gmh = gamma[h];
    const float* wp0 = w + ((size_t)(b * Tt + c * CHUNK)) * Dd + d0;
    const float* wp1 = w + ((size_t)(b * Tt + c * CHUNK)) * Dd + d1;
    float* sp = score + ((size_t)b * NHh + h) * Tt + c * CHUNK;
    const float* pp = penalty + (size_t)b * Tt + c * CHUNK;
    for (int i = 0; i < CHUNK; ++i) {
        float wv0 = wp0[(size_t)i * Dd];
        float wv1 = wp1[(size_t)i * Dd];
        float sq0 = wv0 * wv0, sq1 = wv1 * wv1;
        run0 += sq0; run1 += sq1;
        float wn = sq0 / fmaxf(run0, 1e-12f) + sq1 / fmaxf(run1, 1e-12f);
        for (int o = 16; o; o >>= 1) wn += __shfl_xor(wn, o, 32);
        if (dl == 0) sp[i] = wn * tmph - gmh * pp[i];
    }
}

// ---------------- softmax across heads (axis=1) ----------------
__global__ __launch_bounds__(256) void k_softmax(const float* __restrict__ score,
                                                 float* __restrict__ alpha) {
    const int idx = blockIdx.x * 256 + threadIdx.x;   // b*T + t
    const int b = idx / Tt, t = idx % Tt;
    float s[NHh];
    float mx = -1e30f;
#pragma unroll
    for (int h = 0; h < NHh; ++h) {
        s[h] = score[((size_t)b * NHh + h) * Tt + t];
        mx = fmaxf(mx, s[h]);
    }
    float sum = 0.f;
#pragma unroll
    for (int h = 0; h < NHh; ++h) { s[h] = __expf(s[h] - mx); sum += s[h]; }
    const float inv = 1.f / sum;
#pragma unroll
    for (int h = 0; h < NHh; ++h)
        alpha[((size_t)b * NHh + h) * Tt + t] = s[h] * inv;
}

// ---------------- inclusive cumsum of alpha over T, per (b,h) ----------------
__global__ __launch_bounds__(256) void k_cumalpha(const float* __restrict__ alpha,
                                                  float* __restrict__ cum) {
    const int h = blockIdx.x, b = blockIdx.y;
    const size_t base = ((size_t)b * NHh + h) * Tt + threadIdx.x * 16;
    const int tid = threadIdx.x;
    float v[16];
    float run = 0.f;
#pragma unroll
    for (int i = 0; i < 16; ++i) { float a = alpha[base + i]; run += a; v[i] = run; }
    __shared__ float sums[256];
    sums[tid] = run;
    __syncthreads();
    for (int off = 1; off < 256; off <<= 1) {
        float x = (tid >= off) ? sums[tid - off] : 0.f;
        __syncthreads();
        sums[tid] += x;
        __syncthreads();
    }
    const float exc = sums[tid] - run;
#pragma unroll
    for (int i = 0; i < 16; ++i) cum[base + i] = v[i] + exc;
}

// ---------------- chunk partial sums of w^2 * alpha ----------------
__global__ __launch_bounds__(256) void k_partial_wsqa(const float* __restrict__ w,
                                                      const float* __restrict__ alpha,
                                                      float* __restrict__ part) {
    const int d = blockIdx.x * 256 + threadIdx.x;
    const int c = blockIdx.y, b = blockIdx.z;
    const int h = d >> 6;
    const float* wp = w + ((size_t)(b * Tt + c * CHUNK)) * Dd + d;
    const size_t abase = ((size_t)b * NHh + h) * Tt + c * CHUNK;
    float s = 0.f;
    for (int i = 0; i < CHUNK; ++i) { float v = wp[(size_t)i * Dd]; s += v * v * alpha[abase + i]; }
    part[((size_t)b * NC + c) * Dd + d] = s;
}

// ---------------- final elementwise: attn + y_inner (bf16) ----------------
__global__ __launch_bounds__(256) void k_y(const float* __restrict__ w,
                                           const float* __restrict__ alpha,
                                           const float* __restrict__ cumalpha,
                                           const float* __restrict__ off2,
                                           unsigned short* __restrict__ ybf) {
    const int d = blockIdx.x * 256 + threadIdx.x;
    const int c = blockIdx.y, b = blockIdx.z;
    const int h = d >> 6;
    float run = off2[((size_t)b * NC + c) * Dd + d];
    const float* wp = w + ((size_t)(b * Tt + c * CHUNK)) * Dd + d;
    const size_t abase = ((size_t)b * NHh + h) * Tt + c * CHUNK;
    for (int i = 0; i < CHUNK; ++i) {
        float wv = wp[(size_t)i * Dd];
        float a  = alpha[abase + i];
        run += wv * wv * a;
        float dots = run / (cumalpha[abase + i] + 1e-8f);
        float attn = fminf(1.f / (1.f + dots), 10000.f);
        float yv   = -wv * a * attn;
        ybf[((size_t)(b * Tt + c * CHUNK + i)) * Dd + d] = f2bf(yv);
    }
}

// ---------------- host launcher ----------------
extern "C" void kernel_launch(void* const* d_in, const int* in_sizes, int n_in,
                              void* d_out, int out_size, void* d_ws, size_t ws_size,
                              hipStream_t stream) {
    const float* x      = (const float*)d_in[0];   // (B,T,D)
    const float* phi    = (const float*)d_in[1];   // (B,T)
    const float* W_attn = (const float*)d_in[2];   // (D,D)
    const float* W_proj = (const float*)d_in[3];   // (D,D)
    const float* gamma  = (const float*)d_in[4];   // (NH,1)
    const float* temp   = (const float*)d_in[5];   // (NH,1)

    float* y_out     = (float*)d_out;                      // (B,T,D)
    float* alpha_out = (float*)d_out + (size_t)MM * Dd;    // (B,NH,T)

    char* ws = (char*)d_ws;
    // workspace layout (bytes); xb region is reused for ybf after GEMM1.
    unsigned short* xb   = (unsigned short*)(ws + 0);                    // 32 MiB
    unsigned short* Wat  = (unsigned short*)(ws + 33554432);             //  2 MiB (W_attn^T, bf16)
    unsigned short* Wpt  = (unsigned short*)(ws + 35651584);             //  2 MiB (W_proj^T, bf16)
    float*          wbuf = (float*)        (ws + 37748736);             // 64 MiB
    float*          pen  = (float*)        (ws + 104857600);            // 64 KiB
    float*          par1 = (float*)        (ws + 104923136);            // 256 KiB
    float*          off1 = (float*)        (ws + 105185280);            // 256 KiB
    float*          scor = (float*)        (ws + 105447424);            //  1 MiB
    float*          cuma = (float*)        (ws + 106496000);            //  1 MiB
    float*          par2 = (float*)        (ws + 107544576);            // 256 KiB
    float*          off2 = (float*)        (ws + 107806720);            // 256 KiB
    unsigned short* ybf  = xb;                                          // reuse

    // 1) convert x to bf16; convert+transpose weights to bf16 (KxN)
    k_f32_to_bf16_x4<<<(MM * Dd / 4 + 255) / 256, 256, 0, stream>>>(x, xb, MM * Dd / 4);
    k_transpose_to_bf16<<<dim3(Dd / 32, Dd / 32), 256, 0, stream>>>(W_attn, Wat);
    k_transpose_to_bf16<<<dim3(Dd / 32, Dd / 32), 256, 0, stream>>>(W_proj, Wpt);

    // 2) w = x @ W_attn^T  (f32 accumulate)
    gemm_bf16_nt<<<dim3(Dd / 128, MM / 128), 256, 0, stream>>>(xb, Wat, wbuf, MM);

    // 3) phi penalty
    k_phi<<<Bb, 256, 0, stream>>>(phi, pen);

    // 4) cumsum(w^2): chunk partials + chunk prefix
    k_partial_wsq<<<dim3(Dd / 256, NC, Bb), 256, 0, stream>>>(wbuf, par1);
    k_prefix<<<(Bb * Dd) / 256, 256, 0, stream>>>(par1, off1);

    // 5) score (one wave32 per block, shuffle-only reduction)
    k_score<<<dim3(NC, NHh, Bb), 32, 0, stream>>>(wbuf, off1, pen, gamma, temp, scor);

    // 6) softmax over heads -> alpha (directly into output slot)
    k_softmax<<<(Bb * Tt) / 256, 256, 0, stream>>>(scor, alpha_out);

    // 7) cumsum(alpha)
    k_cumalpha<<<dim3(NHh, Bb), 256, 0, stream>>>(alpha_out, cuma);

    // 8) cumsum(w^2 * alpha): chunk partials + chunk prefix
    k_partial_wsqa<<<dim3(Dd / 256, NC, Bb), 256, 0, stream>>>(wbuf, alpha_out, par2);
    k_prefix<<<(Bb * Dd) / 256, 256, 0, stream>>>(par2, off2);

    // 9) attn + y_inner (bf16, reuses xb region)
    k_y<<<dim3(Dd / 256, NC, Bb), 256, 0, stream>>>(wbuf, alpha_out, cuma, off2, ybf);

    // 10) y = y_inner @ W_proj^T  -> d_out
    gemm_bf16_nt<<<dim3(Dd / 128, MM / 128), 256, 0, stream>>>(ybf, Wpt, y_out, MM);

    (void)in_sizes; (void)n_in; (void)out_size; (void)ws_size;
}
